// GraphAttentionLayer_49890340110388
// MI455X (gfx1250) — matrix operand order repaired and checked by reading, not compiled, not run on hardware
//
#include <hip/hip_runtime.h>
#include <hip/hip_bf16.h>
#include <math.h>

typedef __attribute__((ext_vector_type(16))) _Float16 v16h;
typedef __attribute__((ext_vector_type(8)))  _Float16 v8h;
typedef __attribute__((ext_vector_type(8)))  float    v8f;

#define GB  8          // batch
#define GN  2048       // nodes
#define GIN 128        // in features
#define GOUT 64        // out features
#define GH  4          // heads
#define NEG_SLOPE 0.2f

// ---------------------------------------------------------------------------
// Kernel 1: h = x @ W per (b,head) via WMMA f16->f32.
// Stores hT (o-major f16) for kernel 3's B operand, and attn_l/attn_r = h.a
// Grid: B*H*(N/128) blocks of 256 threads (8 waves, 16 rows each).
// ---------------------------------------------------------------------------
__global__ __launch_bounds__(256) void gat_h_kernel(
    const float* __restrict__ x,       // (B,N,128)
    const float* __restrict__ W,       // (H,128,64)
    const float* __restrict__ aL,      // (H,64)
    const float* __restrict__ aR,      // (H,64)
    _Float16*   __restrict__ hT,       // (B,H,64,N) o-major
    float*      __restrict__ attn_l,   // (B,H,N)
    float*      __restrict__ attn_r)   // (B,H,N)
{
    __shared__ __attribute__((aligned(16))) _Float16 WT[GOUT][GIN + 8]; // transposed W, padded

    const int bid  = blockIdx.x;
    const int tile = bid & 15;         // N/128 tiles
    const int bh   = bid >> 4;
    const int head = bh & (GH - 1);

    // stage W[head] into LDS as f16, transposed (o-major)
    const float* Wh = W + (size_t)head * GIN * GOUT;
    for (int e = threadIdx.x; e < GIN * GOUT; e += 256) {
        int k = e >> 6;                // row (K)
        int o = e & 63;                // col (O)
        WT[o][k] = (_Float16)Wh[e];
    }
    __syncthreads();

    const int wave = threadIdx.x >> 5;
    const int lane = threadIdx.x & 31;
    const int l15  = lane & 15;
    const int half = lane >> 4;

    const int rowbase = tile * 128 + wave * 16;
    const int b       = bh >> 2;
    const float* xrow = x + ((size_t)b * GN + (rowbase + l15)) * GIN;

    v8f acc[4] = {};

    #pragma unroll
    for (int kk = 0; kk < GIN; kk += 32) {
        // ---- A fragment: lane<16 holds K kk+0..15; lane>=16 kk+16..31
        const int kA = kk + (half ? 16 : 0);
        float4 xv[4];
        xv[0] = *(const float4*)(xrow + kA);
        xv[1] = *(const float4*)(xrow + kA + 4);
        xv[2] = *(const float4*)(xrow + kA + 8);
        xv[3] = *(const float4*)(xrow + kA + 12);
        const float* xp = (const float*)xv;
        v16h afrag;
        #pragma unroll
        for (int t = 0; t < 16; ++t) afrag[t] = (_Float16)xp[t];

        // ---- B fragments from LDS: lane<16 holds K kk+0..15; lane>=16 kk+16..31
        const int kB = kk + (half ? 16 : 0);
        #pragma unroll
        for (int nt = 0; nt < 4; ++nt) {
            const _Float16* src = &WT[nt * 16 + l15][kB];
            v8h lo = *(const v8h*)(src);
            v8h hi = *(const v8h*)(src + 8);
            v16h bfrag;
            #pragma unroll
            for (int t = 0; t < 8; ++t) { bfrag[t] = lo[t]; bfrag[t + 8] = hi[t]; }
            acc[nt] = __builtin_amdgcn_wmma_f32_16x16x32_f16(
                false, afrag, false, bfrag, (short)0, acc[nt], false, false);
        }
    }

    // ---- store hT (o-major) and per-row dot products with a_left/a_right
    float pl[8] = {}, pr[8] = {};
    #pragma unroll
    for (int nt = 0; nt < 4; ++nt) {
        const int o = nt * 16 + l15;
        const float al = aL[head * GOUT + o];
        const float ar = aR[head * GOUT + o];
        _Float16* hcol = hT + ((size_t)bh * GOUT + o) * GN;
        #pragma unroll
        for (int v = 0; v < 8; ++v) {
            const int r = rowbase + v + (half ? 8 : 0);
            const float hv = acc[nt][v];
            hcol[r] = (_Float16)hv;
            pl[v] += hv * al;
            pr[v] += hv * ar;
        }
    }
    // reduce across the 16 lanes of each half (rows differ between halves)
    #pragma unroll
    for (int v = 0; v < 8; ++v) {
        #pragma unroll
        for (int m = 8; m >= 1; m >>= 1) {
            pl[v] += __shfl_xor(pl[v], m, 32);
            pr[v] += __shfl_xor(pr[v], m, 32);
        }
    }
    if (l15 == 0) {
        float* lbase = attn_l + (size_t)bh * GN;
        float* rbase = attn_r + (size_t)bh * GN;
        #pragma unroll
        for (int v = 0; v < 8; ++v) {
            const int r = rowbase + v + (half ? 8 : 0);
            lbase[r] = pl[v];
            rbase[r] = pr[v];
        }
    }
}

// ---------------------------------------------------------------------------
// Kernel 2: per adjacency row i (shared by 4 heads): m_i = max_j s_ij,
// invZ_i = 1/sum_j exp(s_ij - m_i).  One wave per row, all 4 heads together.
// Grid: B*N/8 blocks of 256 threads.
// ---------------------------------------------------------------------------
__global__ __launch_bounds__(256) void gat_stats_kernel(
    const int*   __restrict__ adj,     // (B,N,N)
    const float* __restrict__ attn_l,  // (B,H,N)
    const float* __restrict__ attn_r,  // (B,H,N)
    float*       __restrict__ mrow,    // (B,H,N)
    float*       __restrict__ invZ)    // (B,H,N)
{
    const int wave  = threadIdx.x >> 5;
    const int lane  = threadIdx.x & 31;
    const int rowid = blockIdx.x * 8 + wave;    // 0 .. B*N-1
    const int b = rowid >> 11;
    const int i = rowid & (GN - 1);

    const int* arow = adj + ((size_t)b * GN + i) * GN;
    const float* rb = attn_r + (size_t)b * GH * GN;

    float li[GH], mx[GH];
    #pragma unroll
    for (int h = 0; h < GH; ++h) {
        li[h] = attn_l[((size_t)b * GH + h) * GN + i];
        mx[h] = -__builtin_inff();
    }

    // pass 1: masked max (no exp)
    for (int j = lane; j < GN; j += 32) {
        __builtin_prefetch(arow + j + 256, 0, 1);
        const int a = arow[j];
        #pragma unroll
        for (int h = 0; h < GH; ++h) {
            float s = li[h] + rb[h * GN + j];
            s = fmaxf(s, NEG_SLOPE * s);               // LeakyReLU
            mx[h] = a ? fmaxf(mx[h], s) : mx[h];
        }
    }
    #pragma unroll
    for (int h = 0; h < GH; ++h)
        #pragma unroll
        for (int m = 16; m >= 1; m >>= 1)
            mx[h] = fmaxf(mx[h], __shfl_xor(mx[h], m, 32));

    // pass 2: sum of exp (adj row now L2/L0 resident)
    float Z[GH] = {};
    for (int j = lane; j < GN; j += 32) {
        const int a = arow[j];
        #pragma unroll
        for (int h = 0; h < GH; ++h) {
            float s = li[h] + rb[h * GN + j];
            s = fmaxf(s, NEG_SLOPE * s);
            const float p = __expf(s - mx[h]);
            Z[h] += a ? p : 0.0f;
        }
    }
    #pragma unroll
    for (int h = 0; h < GH; ++h)
        #pragma unroll
        for (int m = 16; m >= 1; m >>= 1)
            Z[h] += __shfl_xor(Z[h], m, 32);

    if (lane == 0) {
        #pragma unroll
        for (int h = 0; h < GH; ++h) {
            const size_t idx = ((size_t)b * GH + h) * GN + i;
            mrow[idx] = mx[h];
            invZ[idx] = (Z[h] > 0.0f) ? (1.0f / Z[h]) : 0.0f;
        }
    }
}

// ---------------------------------------------------------------------------
// Kernel 3: out = P @ h per (b,head), FlashAttention-style: P built on the
// fly in the WMMA A-fragment register layout (f16), B from o-major hT.
// Grid: B*H*(N/64) blocks of 128 threads (4 waves, 16 rows each).
// ---------------------------------------------------------------------------
__global__ __launch_bounds__(128) void gat_av_kernel(
    const int*      __restrict__ adj,     // (B,N,N)
    const float*    __restrict__ attn_l,  // (B,H,N)
    const float*    __restrict__ attn_r,  // (B,H,N)
    const _Float16* __restrict__ hT,      // (B,H,64,N)
    const float*    __restrict__ mrow,    // (B,H,N)
    const float*    __restrict__ invZ,    // (B,H,N)
    float*          __restrict__ out)     // (B,N,H*64)
{
    const int bid  = blockIdx.x;
    const int tile = bid & 31;            // N/64 tiles
    const int bh   = bid >> 5;
    const int head = bh & (GH - 1);
    const int b    = bh >> 2;

    const int wave = threadIdx.x >> 5;
    const int lane = threadIdx.x & 31;
    const int l15  = lane & 15;
    const int half = lane >> 4;

    const int ibase = tile * 64 + wave * 16;
    const int myrow = ibase + l15;        // A-fragment row owned by this lane

    const float li = attn_l[(size_t)bh * GN + myrow];
    const float mi = mrow[(size_t)bh * GN + myrow];
    const float zi = invZ[(size_t)bh * GN + myrow];

    const int*      arow = adj + ((size_t)b * GN + myrow) * GN;
    const float*    rv   = attn_r + (size_t)bh * GN;
    const _Float16* hb   = hT + (size_t)bh * GOUT * GN;

    v8f acc[4] = {};

    for (int jb = 0; jb < GN; jb += 32) {
        __builtin_prefetch(arow + jb + 512, 0, 1);
        // ---- A fragment: P_ij as f16.  lane<16: K jb+0..15; lane>=16 jb+16..31
        const int j0 = jb + (half ? 16 : 0);
        int4 aa[4];
        aa[0] = *(const int4*)(arow + j0);
        aa[1] = *(const int4*)(arow + j0 + 4);
        aa[2] = *(const int4*)(arow + j0 + 8);
        aa[3] = *(const int4*)(arow + j0 + 12);
        float4 rr[4];
        rr[0] = *(const float4*)(rv + j0);
        rr[1] = *(const float4*)(rv + j0 + 4);
        rr[2] = *(const float4*)(rv + j0 + 8);
        rr[3] = *(const float4*)(rv + j0 + 12);
        const int*   ap = (const int*)aa;
        const float* rp = (const float*)rr;
        v16h afrag;
        #pragma unroll
        for (int t = 0; t < 16; ++t) {
            float s = li + rp[t];
            s = fmaxf(s, NEG_SLOPE * s);                 // LeakyReLU
            const float p = __expf(s - mi) * zi;         // softmax weight
            afrag[t] = ap[t] ? (_Float16)p : (_Float16)0.0f;
        }

        // ---- B fragments: hT rows o, K(=j) contiguous.  lane<16: K jb+0..15
        const int jB = jb + (half ? 16 : 0);
        #pragma unroll
        for (int nt = 0; nt < 4; ++nt) {
            const _Float16* src = hb + (size_t)(nt * 16 + l15) * GN + jB;
            v8h lo = *(const v8h*)(src);
            v8h hi = *(const v8h*)(src + 8);
            v16h bfrag;
            #pragma unroll
            for (int t = 0; t < 8; ++t) { bfrag[t] = lo[t]; bfrag[t + 8] = hi[t]; }
            acc[nt] = __builtin_amdgcn_wmma_f32_16x16x32_f16(
                false, afrag, false, bfrag, (short)0, acc[nt], false, false);
        }
    }

    // ---- write final output layout (B, N, H*64)
    #pragma unroll
    for (int nt = 0; nt < 4; ++nt) {
        const int o = nt * 16 + l15;
        #pragma unroll
        for (int v = 0; v < 8; ++v) {
            const int r = ibase + v + (half ? 8 : 0);
            out[((size_t)b * GN + r) * (GH * GOUT) + head * GOUT + o] = acc[nt][v];
        }
    }
}

// ---------------------------------------------------------------------------
extern "C" void kernel_launch(void* const* d_in, const int* in_sizes, int n_in,
                              void* d_out, int out_size, void* d_ws, size_t ws_size,
                              hipStream_t stream) {
    const float* x   = (const float*)d_in[0];
    const int*   adj = (const int*)d_in[1];
    const float* W   = (const float*)d_in[2];
    const float* aL  = (const float*)d_in[3];
    const float* aR  = (const float*)d_in[4];
    float* out = (float*)d_out;

    char* ws = (char*)d_ws;
    _Float16* hT = (_Float16*)ws;                                   // 8 MB
    size_t off = (size_t)GB * GH * GOUT * GN * sizeof(_Float16);
    float* attn_l = (float*)(ws + off); off += (size_t)GB * GH * GN * sizeof(float);
    float* attn_r = (float*)(ws + off); off += (size_t)GB * GH * GN * sizeof(float);
    float* mr     = (float*)(ws + off); off += (size_t)GB * GH * GN * sizeof(float);
    float* iZ     = (float*)(ws + off); off += (size_t)GB * GH * GN * sizeof(float);

    gat_h_kernel<<<GB * GH * (GN / 128), 256, 0, stream>>>(x, W, aL, aR, hT, attn_l, attn_r);
    gat_stats_kernel<<<GB * GN / 8, 256, 0, stream>>>(adj, attn_l, attn_r, mr, iZ);
    gat_av_kernel<<<GB * GH * (GN / 64), 128, 0, stream>>>(adj, attn_l, attn_r, hT, mr, iZ, out);
}
